// CRF_19507741458721
// MI455X (gfx1250) — matrix operand order repaired and checked by reading, not compiled, not run on hardware
//
#include <hip/hip_runtime.h>

typedef __attribute__((ext_vector_type(16))) _Float16 v16h;
typedef __attribute__((ext_vector_type(8)))  _Float16 v8h;
typedef __attribute__((ext_vector_type(8)))  float    v8f;
typedef __attribute__((ext_vector_type(4)))  float    v4f;

#define S_LEN   512
#define T_TAGS  128
#define B_BATCH 256

// CDNA5 split workgroup barrier: publish (drain LDS, signal) ... wait.
// Independent global-memory work goes between the two halves.
#define BAR_SIGNAL() asm volatile("s_wait_dscnt 0x0\n\ts_barrier_signal -1" ::: "memory")
#define BAR_WAIT()   asm volatile("s_barrier_wait -1" ::: "memory")

// Raw v_log_f32 (log2); v_nop satisfies the TRANS-op hazard.
__device__ __forceinline__ float fast_log(float x) {
  float r;
  asm("v_log_f32 %0, %1\n\tv_nop" : "=v"(r) : "v"(x));
  return r * 0.6931471805599453f;
}

// clamp to 1e-30 (kills the IR canonicalize max) + 8 in-place v_log_f32.
// Consecutive logs are mutually independent; single v_nop covers the
// 1-op TRANS hazard on the last one.
__device__ __forceinline__ void clamp_log2_x8(float x[8]) {
  asm("v_max_num_f32 %0, 0x0DA24260, %0\n\t"
      "v_max_num_f32 %1, 0x0DA24260, %1\n\t"
      "v_max_num_f32 %2, 0x0DA24260, %2\n\t"
      "v_max_num_f32 %3, 0x0DA24260, %3\n\t"
      "v_max_num_f32 %4, 0x0DA24260, %4\n\t"
      "v_max_num_f32 %5, 0x0DA24260, %5\n\t"
      "v_max_num_f32 %6, 0x0DA24260, %6\n\t"
      "v_max_num_f32 %7, 0x0DA24260, %7\n\t"
      "v_log_f32 %0, %0\n\t"
      "v_log_f32 %1, %1\n\t"
      "v_log_f32 %2, %2\n\t"
      "v_log_f32 %3, %3\n\t"
      "v_log_f32 %4, %4\n\t"
      "v_log_f32 %5, %5\n\t"
      "v_log_f32 %6, %6\n\t"
      "v_log_f32 %7, %7\n\t"
      "v_nop"
      : "+v"(x[0]), "+v"(x[1]), "+v"(x[2]), "+v"(x[3]),
        "+v"(x[4]), "+v"(x[5]), "+v"(x[6]), "+v"(x[7]));
}

// 16-lane xor-max butterfly, two rows interleaved for ILP. Minimal form:
// v_permlane16_b32 (old-value dead under full EXEC) + v_max_num_f32.
__device__ __forceinline__ void rowmax16_x2(float& a, float& b) {
  float t0, t1;
  asm("v_permlane16_b32 %2, %0, %4, %5\n\t"
      "v_permlane16_b32 %3, %1, %4, %5\n\t"
      "v_max_num_f32 %0, %0, %2\n\t"
      "v_max_num_f32 %1, %1, %3\n\t"
      "v_permlane16_b32 %2, %0, %6, %7\n\t"
      "v_permlane16_b32 %3, %1, %6, %7\n\t"
      "v_max_num_f32 %0, %0, %2\n\t"
      "v_max_num_f32 %1, %1, %3\n\t"
      "v_permlane16_b32 %2, %0, %8, %9\n\t"
      "v_permlane16_b32 %3, %1, %8, %9\n\t"
      "v_max_num_f32 %0, %0, %2\n\t"
      "v_max_num_f32 %1, %1, %3\n\t"
      "v_permlane16_b32 %2, %0, %10, %11\n\t"
      "v_permlane16_b32 %3, %1, %10, %11\n\t"
      "v_max_num_f32 %0, %0, %2\n\t"
      "v_max_num_f32 %1, %1, %3"
      : "+v"(a), "+v"(b), "=&v"(t0), "=&v"(t1)
      : "s"((int)0x67452301), "s"((int)0xEFCDAB89),   // xor 1
        "s"((int)0x54761032), "s"((int)0xDCFE98BA),   // xor 2
        "s"((int)0x32107654), "s"((int)0xBA98FEDC),   // xor 4
        "s"((int)0xFEDCBA98), "s"((int)0x76543210));  // xor 8
}

// ---------------------------------------------------------------------------
// prep: c[j] = max_i trans[i,j];  Eh[i,j] = (f16) exp(trans[i,j] - c[j])
// ---------------------------------------------------------------------------
__global__ void crf_prep(const float* __restrict__ trans,
                         _Float16* __restrict__ Eh,
                         float* __restrict__ cmax) {
  int j = threadIdx.x;                      // 128 threads, one column each
  float cm = -3.402823466e38f;
  for (int i = 0; i < T_TAGS; ++i) cm = fmaxf(cm, trans[i * T_TAGS + j]);
  cmax[j] = cm;
  for (int i = 0; i < T_TAGS; ++i)
    Eh[i * T_TAGS + j] = (_Float16)__expf(trans[i * T_TAGS + j] - cm);
}

// ---------------------------------------------------------------------------
// gold score per batch row
// ---------------------------------------------------------------------------
__global__ void crf_gold(const float* __restrict__ emis,
                         const int* __restrict__ tags,
                         const float* __restrict__ trans,
                         float* __restrict__ goldPart) {
  __shared__ float red[256];
  const int b = blockIdx.x, tid = threadIdx.x;
  float acc = 0.f;
  for (int s = tid; s < S_LEN; s += 256) {
    int tg = tags[b * S_LEN + s];
    acc += emis[((size_t)b * S_LEN + s) * T_TAGS + tg];
    if (s + 1 < S_LEN) acc += trans[tg * T_TAGS + tags[b * S_LEN + s + 1]];
  }
  red[tid] = acc;
  __syncthreads();
  for (int off = 128; off > 0; off >>= 1) {
    if (tid < off) red[tid] += red[tid + off];
    __syncthreads();
  }
  if (tid == 0) goldPart[b] = red[0];
}

// ---------------------------------------------------------------------------
// forward scan: 16 blocks x 256 threads (8 wave32s). Block = 16 batch rows.
// alpha register-resident; f16 A-matrix + 16x8 max-exchange in LDS.
// 2 split barriers / step with global loads hidden in the barrier windows.
// ---------------------------------------------------------------------------
__global__ void __launch_bounds__(256)
crf_forward(const float* __restrict__ emis,
            const _Float16* __restrict__ Eh,
            const float* __restrict__ cmax,
            float* __restrict__ logZPart) {
  __shared__ alignas(16) _Float16 aexpH[16][T_TAGS];  // A-matrix, 4 KB
  __shared__ alignas(16) float mred[16][8];           // per-row, per-wave max
  __shared__ float lzsh[16];

  const int tid  = threadIdx.x;
  const int lane = tid & 31;
  const int wid  = tid >> 5;            // 8 waves: wave n owns columns 16n..16n+15
  const int b0   = blockIdx.x * 16;
  const int N0   = wid * 16;
  const int coln = lane & 15;           // D column N0+coln, A row = coln
  const int Md0  = (lane < 16) ? 0 : 8; // D rows Md0..Md0+7
  const int ka   = (lane < 16) ? 0 : 8; // A-matrix K sub-offset per lane half

  // ---- preload constant B fragments (exp(trans-c)), 4 K-tiles of 32x16 ----
  v16h bfrag[4];
  {
    const int kb = (lane < 16) ? 0 : 8;
#pragma unroll
    for (int kt = 0; kt < 4; ++kt)
#pragma unroll
      for (int h = 0; h < 16; ++h)
        bfrag[kt][h] = Eh[(kt * 32 + kb + h + (h >= 8 ? 8 : 0)) * T_TAGS + (N0 + coln)];
  }

  const float cN = cmax[N0 + coln];
  const size_t rowStride = (size_t)S_LEN * T_TAGS;
  const float* ebase = emis + (size_t)b0 * rowStride + (N0 + coln);

  float alpha[8];   // alpha cells (rows Md0..Md0+7, column N0+coln)
  float myM[8];     // current per-row stabilizer m
  float ecur[8];    // emissions for the step being computed
  float enext[8];

  // wave-level per-row max (permlane16 butterfly) + publish to mred
  auto publish_max = [&]() {
    float rmax[8];
#pragma unroll
    for (int r = 0; r < 8; ++r) rmax[r] = alpha[r];
    rowmax16_x2(rmax[0], rmax[1]);
    rowmax16_x2(rmax[2], rmax[3]);
    rowmax16_x2(rmax[4], rmax[5]);
    rowmax16_x2(rmax[6], rmax[7]);
    if ((lane & 15) == 0) {                // single branch: lanes 0/16 publish
#pragma unroll
      for (int r = 0; r < 8; ++r) mred[Md0 + r][wid] = rmax[r];
    }
  };

  // cross-wave max from mred, then store f16 exp(alpha - m) A-matrix
  auto finish_stabilize = [&]() {
#pragma unroll
    for (int r = 0; r < 8; ++r) {
      v4f a = *(const v4f*)&mred[Md0 + r][0];
      v4f b = *(const v4f*)&mred[Md0 + r][4];
      float m0 = fmaxf(fmaxf(a[0], a[1]), fmaxf(a[2], a[3]));
      float m1 = fmaxf(fmaxf(b[0], b[1]), fmaxf(b[2], b[3]));
      myM[r] = fmaxf(m0, m1);
      aexpH[Md0 + r][N0 + coln] = (_Float16)__expf(alpha[r] - myM[r]);
    }
  };

  // ---- prologue: alpha0 = emissions[:, 0, :] ----
#pragma unroll
  for (int r = 0; r < 8; ++r)
    alpha[r] = ebase[(size_t)(Md0 + r) * rowStride];

  publish_max();
  BAR_SIGNAL();
#pragma unroll
  for (int r = 0; r < 8; ++r)            // emissions for t = 1 in the window
    ecur[r] = ebase[(size_t)(Md0 + r) * rowStride + (size_t)T_TAGS];
  BAR_WAIT();
  finish_stabilize();
  BAR_SIGNAL();
  BAR_WAIT();

  for (int t = 1; t < S_LEN; ++t) {
    // A fragments: ds_load_b128 pairs overlap with preceding WMMA execution
    v16h afr[4];
#pragma unroll
    for (int kt = 0; kt < 4; ++kt) {
      v8h lo = *(const v8h*)&aexpH[coln][kt * 32 + ka];
      v8h hi = *(const v8h*)&aexpH[coln][kt * 32 + ka + 16];
#pragma unroll
      for (int h = 0; h < 8; ++h) { afr[kt][h] = lo[h]; afr[kt][h + 8] = hi[h]; }
    }

    // s = exp(alpha-m) @ exp(trans-c): 4 chained WMMAs, K = 128, f32 acc
    v8f acc = {};
#pragma unroll
    for (int kt = 0; kt < 4; ++kt)
      acc = __builtin_amdgcn_wmma_f32_16x16x32_f16(
              false, afr[kt], false, bfrag[kt], (short)0, acc, false, false);

    // alpha_new = m + c_j + log(s) + e_t   (stays in registers)
    float lg[8];
#pragma unroll
    for (int r = 0; r < 8; ++r) lg[r] = acc[r];
    clamp_log2_x8(lg);
#pragma unroll
    for (int r = 0; r < 8; ++r)
      alpha[r] = fmaf(lg[r], 0.6931471805599453f, myM[r] + cN + ecur[r]);

    publish_max();
    BAR_SIGNAL();
    // --- barrier window: independent global traffic for future steps ---
    {
      const int tn = (t + 1 < S_LEN) ? (t + 1) : (S_LEN - 1);
#pragma unroll
      for (int r = 0; r < 8; ++r)
        enext[r] = ebase[(size_t)(Md0 + r) * rowStride + (size_t)tn * T_TAGS];
      if (t + 2 < S_LEN)
        __builtin_prefetch(
            ebase + (size_t)Md0 * rowStride + (size_t)(t + 2) * T_TAGS, 0, 3);
    }
    BAR_WAIT();

    finish_stabilize();
    BAR_SIGNAL();
#pragma unroll
    for (int r = 0; r < 8; ++r) ecur[r] = enext[r];   // register moves only
    BAR_WAIT();
  }

  // ---- final logsumexp per batch row: reuse aexpH = exp(alpha-m), mred ----
  if (tid < 16) {
    float sum = 0.f;
    const _Float16* row = &aexpH[tid][0];
    for (int c = 0; c < T_TAGS; ++c) sum += (float)row[c];
    v4f a = *(const v4f*)&mred[tid][0];
    v4f b = *(const v4f*)&mred[tid][4];
    float mx = fmaxf(fmaxf(fmaxf(a[0], a[1]), fmaxf(a[2], a[3])),
                     fmaxf(fmaxf(b[0], b[1]), fmaxf(b[2], b[3])));
    lzsh[tid] = mx + fast_log(sum);
  }
  __syncthreads();
  if (tid == 0) {
    float z = 0.f;
#pragma unroll
    for (int r = 0; r < 16; ++r) z += lzsh[r];
    logZPart[blockIdx.x] = z;
  }
}

// ---------------------------------------------------------------------------
// final combine: out = mean(logZ) - mean(gold)
// ---------------------------------------------------------------------------
__global__ void crf_reduce(const float* __restrict__ goldPart,
                           const float* __restrict__ logZPart,
                           float* __restrict__ out) {
  __shared__ float red[256];
  const int tid = threadIdx.x;
  red[tid] = goldPart[tid];
  __syncthreads();
  for (int off = 128; off > 0; off >>= 1) {
    if (tid < off) red[tid] += red[tid + off];
    __syncthreads();
  }
  if (tid == 0) {
    float z = 0.f;
    for (int i = 0; i < 16; ++i) z += logZPart[i];
    out[0] = (z - red[0]) / (float)B_BATCH;
  }
}

// ---------------------------------------------------------------------------
extern "C" void kernel_launch(void* const* d_in, const int* in_sizes, int n_in,
                              void* d_out, int out_size, void* d_ws, size_t ws_size,
                              hipStream_t stream) {
  const float* emis  = (const float*)d_in[0];   // (256, 512, 128) f32
  const int*   tags  = (const int*)d_in[1];     // (256, 512) int
  const float* trans = (const float*)d_in[2];   // (128, 128) f32
  float* out = (float*)d_out;                   // scalar f32

  char* ws = (char*)d_ws;
  _Float16* Eh    = (_Float16*)ws;              // 128*128 f16 = 32768 B
  float* cmaxp    = (float*)(ws + 32768);       // 128 f32
  float* goldPart = (float*)(ws + 33280);       // 256 f32
  float* logZPart = (float*)(ws + 34304);       // 16 f32

  crf_prep   <<<1,       128, 0, stream>>>(trans, Eh, cmaxp);
  crf_gold   <<<B_BATCH, 256, 0, stream>>>(emis, tags, trans, goldPart);
  crf_forward<<<16,      256, 0, stream>>>(emis, Eh, cmaxp, logZPart);
  crf_reduce <<<1,       256, 0, stream>>>(goldPart, logZPart, out);
}
